// LinearFLH_85658827751871
// MI455X (gfx1250) — matrix operand order repaired and checked by reading, not compiled, not run on hardware
//
#include <hip/hip_runtime.h>
#include <stdint.h>

typedef __attribute__((ext_vector_type(16))) _Float16 v16h;
typedef __attribute__((ext_vector_type(8)))  _Float16 v8h;
typedef __attribute__((ext_vector_type(8)))  float    v8f;
typedef __attribute__((ext_vector_type(4)))  int      v4i;

// typed pointers in explicit address spaces for the async-LDS builtin:
// arg0 = global (AS1) v4i*, arg1 = LDS (AS3) v4i*  (signature revealed by clang diagnostics)
typedef __attribute__((address_space(1))) v4i* gv4i_p;
typedef __attribute__((address_space(3))) v4i* lv4i_p;

#define M_DIM 8192
#define N_DIM 11008
#define K_DIM 4096
#define KT    64          // K-tile per LDS stage
#define LDSW  72          // padded LDS row stride in halves (144 B, 16B-aligned)

#if defined(__has_builtin)
#if __has_builtin(__builtin_amdgcn_global_load_async_to_lds_b128)
#define HAVE_ASYNC_LDS 1
#endif
#if __has_builtin(__builtin_amdgcn_s_wait_asynccnt)
#define HAVE_WAIT_ASYNC_BUILTIN 1
#endif
#endif
#ifndef HAVE_ASYNC_LDS
#define HAVE_ASYNC_LDS 0
#endif

// flat -> AS pointer casts via integer round-trip:
// global flat addresses equal AS1 addresses; flat LDS addr[31:0] is the LDS offset.
#define AS1(p) ((gv4i_p)(uintptr_t)(p))
#define AS3(p) ((lv4i_p)(uintptr_t)(p))

__device__ __forceinline__ void wait_async0() {
#if defined(HAVE_WAIT_ASYNC_BUILTIN)
    __builtin_amdgcn_s_wait_asynccnt(0);
#else
    asm volatile("s_wait_asynccnt 0x0" ::: "memory");
#endif
}

__global__ __launch_bounds__(256)
void qlinear_wmma_f16(const _Float16* __restrict__ xq,   // [8192, 4096]
                      const _Float16* __restrict__ sx,   // [8192]
                      const _Float16* __restrict__ wt,   // [11008, 4096]
                      const _Float16* __restrict__ ws,   // [11008]
                      const _Float16* __restrict__ bias, // [11008]
                      float* __restrict__ out)           // [8192, 11008]
{
    __shared__ __attribute__((aligned(16))) _Float16 As[2][128][LDSW];
    __shared__ __attribute__((aligned(16))) _Float16 Bs[2][128][LDSW];

    const int tid   = threadIdx.x;
    const int lane  = tid & 31;
    const int wave  = tid >> 5;
    const int waveM = wave >> 2;   // 0..1 (64 rows each)
    const int waveN = wave & 3;    // 0..3 (32 cols each)
    const int hl    = lane >> 4;   // half-wave select
    const int l16   = lane & 15;

    const int mBlock = blockIdx.y * 128;
    const int nBlock = blockIdx.x * 128;

    v8f acc[4][2];
    {
        v8f z = {0.f, 0.f, 0.f, 0.f, 0.f, 0.f, 0.f, 0.f};
#pragma unroll
        for (int i = 0; i < 4; ++i)
#pragma unroll
            for (int j = 0; j < 2; ++j) acc[i][j] = z;
    }

    // ---- stage K-tile 0 into buffer 0 ----
#pragma unroll
    for (int t = 0; t < 4; ++t) {
        const int c   = t * 256 + tid;        // 1024 b128 chunks per tile
        const int row = c >> 3;
        const int seg = c & 7;
        const _Float16* ga = xq + (size_t)(mBlock + row) * K_DIM + seg * 8;
        const _Float16* gb = wt + (size_t)(nBlock + row) * K_DIM + seg * 8;
        _Float16* la = &As[0][row][seg * 8];
        _Float16* lb = &Bs[0][row][seg * 8];
#if HAVE_ASYNC_LDS
        __builtin_amdgcn_global_load_async_to_lds_b128(AS1(ga), AS3(la), 0, 0);
        __builtin_amdgcn_global_load_async_to_lds_b128(AS1(gb), AS3(lb), 0, 0);
#else
        *(v8h*)la = *(const v8h*)ga;
        *(v8h*)lb = *(const v8h*)gb;
#endif
    }
#if HAVE_ASYNC_LDS
    wait_async0();
#endif
    __syncthreads();

    const int NT = K_DIM / KT;   // 64 k-steps
    for (int kt = 0; kt < NT; ++kt) {
        const int  buf      = kt & 1;
        const bool haveNext = (kt + 1 < NT);

        // ---- start fill of next tile (overlaps with WMMA below) ----
#if !HAVE_ASYNC_LDS
        v8h ar[4], br[4];
#endif
        if (haveNext) {
            const int kb = (kt + 1) * KT;
            const int nb = buf ^ 1;
            (void)nb;
#pragma unroll
            for (int t = 0; t < 4; ++t) {
                const int c   = t * 256 + tid;
                const int row = c >> 3;
                const int seg = c & 7;
                const _Float16* ga = xq + (size_t)(mBlock + row) * K_DIM + kb + seg * 8;
                const _Float16* gb = wt + (size_t)(nBlock + row) * K_DIM + kb + seg * 8;
#if HAVE_ASYNC_LDS
                __builtin_amdgcn_global_load_async_to_lds_b128(
                    AS1(ga), AS3(&As[nb][row][seg * 8]), 0, 0);
                __builtin_amdgcn_global_load_async_to_lds_b128(
                    AS1(gb), AS3(&Bs[nb][row][seg * 8]), 0, 0);
#else
                ar[t] = *(const v8h*)ga;
                br[t] = *(const v8h*)gb;
#endif
            }
            if (kt + 2 < NT) {           // global_prefetch_b8 two tiles ahead
                const int kb2 = (kt + 2) * KT;
                const int row = tid >> 3, seg = tid & 7;
                __builtin_prefetch(xq + (size_t)(mBlock + row) * K_DIM + kb2 + seg * 8, 0, 1);
                __builtin_prefetch(wt + (size_t)(nBlock + row) * K_DIM + kb2 + seg * 8, 0, 1);
            }
        }

        // ---- compute on current buffer: 2 k-subtiles of 32, 16 WMMAs total ----
#pragma unroll
        for (int ks = 0; ks < KT; ks += 32) {
            union { v16h v; v8h h[2]; } bfr[2];
#pragma unroll
            for (int wn = 0; wn < 2; ++wn) {
                const int n = waveN * 32 + wn * 16 + l16;
                const int k = ks + hl * 8;      // lane<16: K0..7,16..23, lane>=16: K8..15,24..31
                bfr[wn].h[0] = *(const v8h*)&Bs[buf][n][k];
                bfr[wn].h[1] = *(const v8h*)&Bs[buf][n][k + 16];
            }
#pragma unroll
            for (int wm = 0; wm < 4; ++wm) {
                union { v16h v; v8h h[2]; } afr;
                const int m = waveM * 64 + wm * 16 + l16;
                afr.h[0] = *(const v8h*)&As[buf][m][ks + hl * 8];        // K 0..7 / 8..15
                afr.h[1] = *(const v8h*)&As[buf][m][ks + 16 + hl * 8];   // K 16..23 / 24..31
#pragma unroll
                for (int wn = 0; wn < 2; ++wn) {
                    acc[wm][wn] = __builtin_amdgcn_wmma_f32_16x16x32_f16(
                        false, afr.v, false, bfr[wn].v,
                        (short)0, acc[wm][wn], false, false);
                }
            }
        }

        // ---- make next tile visible before the next iteration ----
        if (haveNext) {
#if HAVE_ASYNC_LDS
            wait_async0();
#else
            const int nb = buf ^ 1;
#pragma unroll
            for (int t = 0; t < 4; ++t) {
                const int c   = t * 256 + tid;
                const int row = c >> 3;
                const int seg = c & 7;
                *(v8h*)&As[nb][row][seg * 8] = ar[t];
                *(v8h*)&Bs[nb][row][seg * 8] = br[t];
            }
#endif
        }
        __syncthreads();
    }

    // ---- epilogue: out[m,n] = acc * sx[m] * ws[n] + bias[n] ----
#pragma unroll
    for (int wn = 0; wn < 2; ++wn) {
        const int   n   = nBlock + waveN * 32 + wn * 16 + l16;
        const float wsc = (float)ws[n];
        const float bv  = (float)bias[n];
#pragma unroll
        for (int wm = 0; wm < 4; ++wm) {
            const int mBase = mBlock + waveM * 64 + wm * 16 + hl * 8;
#pragma unroll
            for (int r = 0; r < 8; ++r) {
                const int   m = mBase + r;
                const float s = (float)sx[m];
                out[(size_t)m * N_DIM + n] = acc[wm][wn][r] * s * wsc + bv;
            }
        }
    }
}

extern "C" void kernel_launch(void* const* d_in, const int* in_sizes, int n_in,
                              void* d_out, int out_size, void* d_ws, size_t ws_size,
                              hipStream_t stream) {
    (void)in_sizes; (void)n_in; (void)out_size; (void)d_ws; (void)ws_size;
    const _Float16* xq   = (const _Float16*)d_in[0]; // quantized_x  [4,2048,4096]
    const _Float16* sx   = (const _Float16*)d_in[1]; // scales_x     [4,2048,1]
    const _Float16* wt   = (const _Float16*)d_in[2]; // weight       [11008,4096]
    const _Float16* ws   = (const _Float16*)d_in[3]; // weight_scales[11008,1]
    const _Float16* bias = (const _Float16*)d_in[4]; // bias         [11008]
    float* out = (float*)d_out;

    dim3 grid(N_DIM / 128, M_DIM / 128);  // 86 x 64 blocks
    qlinear_wmma_f16<<<grid, 256, 0, stream>>>(xq, sx, wt, ws, bias, out);
}
